// ModulationAttn2d_8065948582192
// MI455X (gfx1250) — matrix operand order repaired and checked by reading, not compiled, not run on hardware
//
#include <hip/hip_runtime.h>
#include <hip/hip_bf16.h>

// ---------------------------------------------------------------------------
// ModulationAttn2d for MI455X (gfx1250, wave32, WMMA + async global->LDS)
// B=8, Cin=256, Cout=256, H=W=64, L=4096
// ---------------------------------------------------------------------------

typedef __attribute__((ext_vector_type(16))) __bf16 v16bf;
typedef __attribute__((ext_vector_type(8)))  __bf16 v8bf;
typedef __attribute__((ext_vector_type(8)))  float  v8f;

#define BQ 8
#define CIN 256
#define LL 4096
#define BSTRIDE 40   // padded k-stride of transposed B tile (80B row: aligned)

// ------------------- async global->LDS (ASYNCcnt) helpers -------------------

#if __has_builtin(__builtin_amdgcn_global_load_async_to_lds_b128)
#define HAVE_ASYNC_LDS 1
typedef int v4i __attribute__((vector_size(16)));
typedef __attribute__((address_space(1))) v4i gv4i;   // global
typedef __attribute__((address_space(3))) v4i sv4i;   // LDS
__device__ __forceinline__ void async_copy16(const __bf16* g, __bf16* l) {
  __builtin_amdgcn_global_load_async_to_lds_b128((gv4i*)g, (sv4i*)l, 0, 0);
}
#else
#define HAVE_ASYNC_LDS 0
__device__ __forceinline__ void async_copy16(const __bf16* g, __bf16* l) {
  *(v8bf*)l = *(const v8bf*)g;   // fallback: through VGPRs
}
#endif

#if __has_builtin(__builtin_amdgcn_s_wait_asynccnt)
#define WAIT_ASYNC() __builtin_amdgcn_s_wait_asynccnt(0)
#elif HAVE_ASYNC_LDS
#define WAIT_ASYNC() asm volatile("s_wait_asynccnt 0" ::: "memory")
#else
#define WAIT_ASYNC() ((void)0)
#endif

// ---------------- WMMA helpers (layouts per CDNA5 ISA 7.12.2) ----------------

__device__ __forceinline__ v8f wmma_bf16(v16bf a, v16bf b, v8f c) {
  return __builtin_amdgcn_wmma_f32_16x16x32_bf16(
      false, a, false, b, (short)0, c, false, false);
}

__device__ __forceinline__ v16bf load_afrag(const __bf16* As, int lane) {
  const int m = lane & 15, h = lane >> 4;
  const v8bf lo = *(const v8bf*)(As + m * 32 + h * 8);
  const v8bf hi = *(const v8bf*)(As + m * 32 + 16 + h * 8);
  v16bf a;
#pragma unroll
  for (int i = 0; i < 8; ++i) { a[i] = lo[i]; a[i + 8] = hi[i]; }
  return a;
}

__device__ __forceinline__ v16bf load_bfrag(const __bf16* Bs, int lane) {
  const int n = lane & 15, h = lane >> 4;
  const v8bf lo = *(const v8bf*)(Bs + n * BSTRIDE + h * 8);
  const v8bf hi = *(const v8bf*)(Bs + n * BSTRIDE + 16 + h * 8);
  v16bf b;
#pragma unroll
  for (int i = 0; i < 8; ++i) { b[i] = lo[i]; b[i + 8] = hi[i]; }
  return b;
}

// ------------------------------ cast kernels -------------------------------

__global__ void k_cast_bf16(const float* __restrict__ in, __bf16* __restrict__ out, int n) {
  int i = blockIdx.x * 256 + threadIdx.x;
  if (i < n) out[i] = (__bf16)in[i];
}

__global__ void k_cast_wqkv(const float* __restrict__ w, __bf16* __restrict__ out) {
  int i = blockIdx.x * 256 + threadIdx.x;
  if (i >= 768 * 256 * 9) return;
  int tap = i % 9, ci = (i / 9) % 256, co = i / (9 * 256);
  out[((size_t)tap * 768 + co) * 256 + ci] = (__bf16)w[i];
}

__global__ void k_cast_wmod(const float* __restrict__ w, __bf16* __restrict__ out) {
  int i = blockIdx.x * 256 + threadIdx.x;
  if (i >= 512 * 256 * 3) return;
  int tap = i % 3, ci = (i / 3) % 256, co = i / (3 * 256);
  out[((size_t)tap * 512 + co) * 256 + ci] = (__bf16)w[i];
}

// --------------------- qkv: 3x3 conv as implicit GEMM ----------------------
// M=768, N=4096, K=9*256. grid(64 h-rows, 12 cout tiles, 8), block 256.
// Tap loop fully unrolled: dx/dy compile-time per copy, no div/mod chains.
__global__ void k_conv_qkv(const __bf16* __restrict__ xb,  // [B][256][64][64]
                           const __bf16* __restrict__ wt,  // [9][768][256]
                           float* __restrict__ q, float* __restrict__ k,
                           __bf16* __restrict__ vb) {
  __shared__ __align__(16) __bf16 As[64 * 32];
  __shared__ __align__(16) __bf16 BsRaw[66 * BSTRIDE];  // halo row above & below
  __bf16* Bs = BsRaw + BSTRIDE;                         // rows -1..64 valid
  const int t = threadIdx.x, lane = t & 31, wav = t >> 5;
  const int wm = wav >> 1, wn = wav & 1;
  const int hrow = blockIdx.x, cout0 = blockIdx.y * 64, b = blockIdx.z;
  const __bf16* xbatch = xb + (size_t)b * CIN * LL;
  const int mA = t >> 2, kA = (t & 3) * 8;
  const int kkB = t >> 3, nbB = (t & 7) * 8;

  auto loadB = [&](int tap, int cc) -> v8bf {  // aligned, unshifted row chunk
    int hs = hrow + tap / 3 - 1;
    v8bf z = {};
    if ((unsigned)hs >= 64u) return z;
    return *(const v8bf*)(xbatch + ((size_t)((cc << 5) + kkB) << 12) + (hs << 6) + nbB);
  };

  v8bf bReg = loadB(0, 0);
  v8f acc0 = {}, acc1 = {};
#pragma unroll
  for (int tap = 0; tap < 9; ++tap) {
    const int dx = tap % 3 - 1;   // compile-time constant (tap unrolled)
    for (int cc = 0; cc < 8; ++cc) {
      __syncthreads();
      async_copy16(wt + ((size_t)tap * 768 + cout0 + mA) * 256 + (cc << 5) + kA,
                   As + mA * 32 + kA);
#pragma unroll
      for (int i = 0; i < 8; ++i)                 // unconditional: halo rows absorb
        Bs[(nbB + i - dx) * BSTRIDE + kkB] = bReg[i];
      if (dx != 0 && t < 32)                      // out-of-image halo column = 0
        Bs[(dx == 1 ? 63 : 0) * BSTRIDE + t] = (__bf16)0.0f;
      WAIT_ASYNC();
      __syncthreads();
      int ncc = cc + 1, ntap = tap;
      if (ncc == 8) { ncc = 0; ++ntap; }
      if (ntap < 9) bReg = loadB(ntap, ncc);
      v16bf af = load_afrag(As + wm * 16 * 32, lane);
      acc0 = wmma_bf16(af, load_bfrag(Bs + (wn * 32) * BSTRIDE, lane), acc0);
      acc1 = wmma_bf16(af, load_bfrag(Bs + (wn * 32 + 16) * BSTRIDE, lane), acc1);
    }
  }
  const int h = lane >> 4, nI = lane & 15;
#pragma unroll
  for (int j = 0; j < 2; ++j) {
    v8f a = j ? acc1 : acc0;
    const int l = (hrow << 6) + wn * 32 + j * 16 + nI;
#pragma unroll
    for (int r = 0; r < 8; ++r) {
      int co = (cout0 & 255) + wm * 16 + h * 8 + r;
      size_t off = ((size_t)b * 256 + co) * LL + l;
      if (cout0 < 256) q[off] = a[r];
      else if (cout0 < 512) k[off] = a[r];
      else vb[off] = (__bf16)a[r];
    }
  }
}

// --------------------- mod: 1-D conv (k=3) as implicit GEMM -----------------
// M=512, N=4096, K=3*256. grid(64 l-tiles, 8 cout tiles, 8).
__global__ void k_conv_mod(const __bf16* __restrict__ vb,  // [B][256][4096]
                           const __bf16* __restrict__ wt,  // [3][512][256]
                           float* __restrict__ qm, float* __restrict__ km) {
  __shared__ __align__(16) __bf16 As[64 * 32];
  __shared__ __align__(16) __bf16 BsRaw[66 * BSTRIDE];
  __bf16* Bs = BsRaw + BSTRIDE;
  const int t = threadIdx.x, lane = t & 31, wav = t >> 5;
  const int wm = wav >> 1, wn = wav & 1;
  const int l0 = blockIdx.x * 64, cout0 = blockIdx.y * 64, b = blockIdx.z;
  const __bf16* vbatch = vb + (size_t)b * CIN * LL;
  const int mA = t >> 2, kA = (t & 3) * 8;
  const int kkB = t >> 3, nbB = (t & 7) * 8;

  auto loadB = [&](int cc) -> v8bf {
    return *(const v8bf*)(vbatch + ((size_t)((cc << 5) + kkB) << 12) + l0 + nbB);
  };

  v8bf bReg = loadB(0);
  v8f acc0 = {}, acc1 = {};
#pragma unroll
  for (int tap = 0; tap < 3; ++tap) {
    const int dt = tap - 1;       // compile-time constant (tap unrolled)
    for (int cc = 0; cc < 8; ++cc) {
      const int c0 = cc << 5;
      __syncthreads();
      async_copy16(wt + ((size_t)tap * 512 + cout0 + mA) * 256 + c0 + kA,
                   As + mA * 32 + kA);
#pragma unroll
      for (int i = 0; i < 8; ++i)
        Bs[(nbB + i - dt) * BSTRIDE + kkB] = bReg[i];
      if (dt != 0 && t < 32) {                    // halo column crosses tile edge
        int nm = (dt == 1) ? 63 : 0;
        int lm = l0 + nm + dt;
        __bf16 hv = (__bf16)0.0f;
        if ((unsigned)lm < (unsigned)LL) hv = vbatch[((size_t)(c0 + t) << 12) + lm];
        Bs[nm * BSTRIDE + t] = hv;
      }
      WAIT_ASYNC();
      __syncthreads();
      int ncc = cc + 1, ntap = tap;
      if (ncc == 8) { ncc = 0; ++ntap; }
      if (ntap < 3) bReg = loadB(ncc);
      v16bf af = load_afrag(As + wm * 16 * 32, lane);
      acc0 = wmma_bf16(af, load_bfrag(Bs + (wn * 32) * BSTRIDE, lane), acc0);
      acc1 = wmma_bf16(af, load_bfrag(Bs + (wn * 32 + 16) * BSTRIDE, lane), acc1);
    }
  }
  const int h = lane >> 4, nI = lane & 15;
  float* dst = (cout0 < 256) ? qm : km;
#pragma unroll
  for (int j = 0; j < 2; ++j) {
    v8f a = j ? acc1 : acc0;
    const int l = l0 + wn * 32 + j * 16 + nI;
#pragma unroll
    for (int r = 0; r < 8; ++r) {
      int co = (cout0 & 255) + wm * 16 + h * 8 + r;
      dst[((size_t)b * 256 + co) * LL + l] = a[r];
    }
  }
}

// ---------------- fused modulate + row softmax over L=4096 ------------------
__global__ void k_mod_softmax(const float* __restrict__ val, const float* __restrict__ mod,
                              __bf16* __restrict__ outb) {
  __shared__ float red[256];
  const int row = blockIdx.x;
  const float* pv = val + (size_t)row * LL;
  const float* pm = mod + (size_t)row * LL;
  float tl[16];
  float mx = -1e30f;
#pragma unroll
  for (int e = 0; e < 16; ++e) {
    int i = threadIdx.x + (e << 8);
    float p = pv[i] * pm[i];
    float tv = -(p > 0.0f ? p : 0.0f);
    tl[e] = tv;
    mx = fmaxf(mx, tv);
  }
  red[threadIdx.x] = mx;
  __syncthreads();
  for (int s = 128; s > 0; s >>= 1) {
    if (threadIdx.x < s) red[threadIdx.x] = fmaxf(red[threadIdx.x], red[threadIdx.x + s]);
    __syncthreads();
  }
  mx = red[0];
  __syncthreads();
  float sum = 0.0f;
#pragma unroll
  for (int e = 0; e < 16; ++e) { tl[e] = __expf(tl[e] - mx); sum += tl[e]; }
  red[threadIdx.x] = sum;
  __syncthreads();
  for (int s = 128; s > 0; s >>= 1) {
    if (threadIdx.x < s) red[threadIdx.x] += red[threadIdx.x + s];
    __syncthreads();
  }
  float inv = 1.0f / red[0];
#pragma unroll
  for (int e = 0; e < 16; ++e) {
    int i = threadIdx.x + (e << 8);
    outb[(size_t)row * LL + i] = (__bf16)(tl[e] * inv);
  }
}

// ------------------ S = m_q * m_k^T * scale  (256x256, K=4096) --------------
// grid(4 d-tiles, 4 c-tiles, 8). Both operands k-contiguous in memory:
// fully async-staged, double-buffered LDS, one barrier per K-step.
__global__ void k_attn_scores(const __bf16* __restrict__ mq, const __bf16* __restrict__ mk,
                              float* __restrict__ S) {
  __shared__ __align__(16) __bf16 As[2][64 * 32];
  __shared__ __align__(16) __bf16 Bs[2][64 * BSTRIDE];
  const int t = threadIdx.x, lane = t & 31, wav = t >> 5;
  const int wm = wav >> 1, wn = wav & 1;
  const int d0 = blockIdx.x * 64, c0 = blockIdx.y * 64, b = blockIdx.z;
  const int mA = t >> 2, kA = (t & 3) * 8;
  const __bf16* pq = mq + (size_t)b * 256 * LL + (size_t)(c0 + mA) * LL + kA;
  const __bf16* pk = mk + (size_t)b * 256 * LL + (size_t)(d0 + mA) * LL + kA;
  v8f acc0 = {}, acc1 = {};
  async_copy16(pq, &As[0][mA * 32 + kA]);
  async_copy16(pk, &Bs[0][mA * BSTRIDE + kA]);
  for (int k0 = 0; k0 < LL; k0 += 32) {
    const int cur = (k0 >> 5) & 1;
    WAIT_ASYNC();          // copies into buf[cur] (issued last iter) complete
    __syncthreads();       // all threads done waiting & done reading buf[cur^1]
    const int kn = k0 + 32;
    if (kn < LL) {
      async_copy16(pq + kn, &As[cur ^ 1][mA * 32 + kA]);
      async_copy16(pk + kn, &Bs[cur ^ 1][mA * BSTRIDE + kA]);
    }
    v16bf af = load_afrag(&As[cur][wm * 16 * 32], lane);
    acc0 = wmma_bf16(af, load_bfrag(&Bs[cur][(wn * 32) * BSTRIDE], lane), acc0);
    acc1 = wmma_bf16(af, load_bfrag(&Bs[cur][(wn * 32 + 16) * BSTRIDE], lane), acc1);
  }
  const float scale = 0.0625f;  // 1/sqrt(256)
  const int h = lane >> 4, nI = lane & 15;
#pragma unroll
  for (int j = 0; j < 2; ++j) {
    v8f a = j ? acc1 : acc0;
    const int d = d0 + wn * 32 + j * 16 + nI;
#pragma unroll
    for (int r = 0; r < 8; ++r) {
      int c = c0 + wm * 16 + h * 8 + r;
      S[((size_t)b * 256 + c) * 256 + d] = a[r] * scale;
    }
  }
}

// ----------------------- softmax over d (rows of 256) -----------------------
__global__ void k_attn_softmax(const float* __restrict__ S, __bf16* __restrict__ mab) {
  __shared__ float red[256];
  const int row = blockIdx.x;
  float vv = S[(size_t)row * 256 + threadIdx.x];
  red[threadIdx.x] = vv;
  __syncthreads();
  for (int s = 128; s > 0; s >>= 1) {
    if (threadIdx.x < s) red[threadIdx.x] = fmaxf(red[threadIdx.x], red[threadIdx.x + s]);
    __syncthreads();
  }
  float mx = red[0];
  __syncthreads();
  float ex = __expf(vv - mx);
  red[threadIdx.x] = ex;
  __syncthreads();
  for (int s = 128; s > 0; s >>= 1) {
    if (threadIdx.x < s) red[threadIdx.x] += red[threadIdx.x + s];
    __syncthreads();
  }
  mab[(size_t)row * 256 + threadIdx.x] = (__bf16)(ex / red[0]);
}

// ------------------------- y = m_attn * v (K=256) ---------------------------
// grid(64 l-tiles, 4 c-tiles, 8)
__global__ void k_attn_apply(const __bf16* __restrict__ mab, const __bf16* __restrict__ vb,
                             float* __restrict__ y, __bf16* __restrict__ yb) {
  __shared__ __align__(16) __bf16 As[64 * 32];
  __shared__ __align__(16) __bf16 BsRaw[66 * BSTRIDE];
  __bf16* Bs = BsRaw + BSTRIDE;
  const int t = threadIdx.x, lane = t & 31, wav = t >> 5;
  const int wm = wav >> 1, wn = wav & 1;
  const int l0 = blockIdx.x * 64, c0 = blockIdx.y * 64, b = blockIdx.z;
  const __bf16* pa = mab + (size_t)b * 256 * 256;
  const __bf16* pv = vb + (size_t)b * 256 * LL;
  const int mA = t >> 2, kA = (t & 3) * 8;
  const int kkB = t >> 3, nbB = (t & 7) * 8;
  v8f acc0 = {}, acc1 = {};
  v8bf bReg = *(const v8bf*)(pv + ((size_t)kkB << 12) + l0 + nbB);
  for (int k0 = 0; k0 < 256; k0 += 32) {
    __syncthreads();
    async_copy16(pa + (size_t)(c0 + mA) * 256 + k0 + kA, As + mA * 32 + kA);
#pragma unroll
    for (int i = 0; i < 8; ++i) Bs[(nbB + i) * BSTRIDE + kkB] = bReg[i];
    WAIT_ASYNC();
    __syncthreads();
    const int kn = k0 + 32;
    if (kn < 256)
      bReg = *(const v8bf*)(pv + ((size_t)(kn + kkB) << 12) + l0 + nbB);
    v16bf af = load_afrag(As + wm * 16 * 32, lane);
    acc0 = wmma_bf16(af, load_bfrag(Bs + (wn * 32) * BSTRIDE, lane), acc0);
    acc1 = wmma_bf16(af, load_bfrag(Bs + (wn * 32 + 16) * BSTRIDE, lane), acc1);
  }
  const int h = lane >> 4, nI = lane & 15;
#pragma unroll
  for (int j = 0; j < 2; ++j) {
    v8f a = j ? acc1 : acc0;
    const int l = l0 + wn * 32 + j * 16 + nI;
#pragma unroll
    for (int r = 0; r < 8; ++r) {
      int c = c0 + wm * 16 + h * 8 + r;
      size_t off = ((size_t)b * 256 + c) * LL + l;
      y[off] = a[r];
      yb[off] = (__bf16)a[r];
    }
  }
}

// --------- final: g = sigmoid(Wg*[y;x]); out = g*(Wr*x) + (1-g)*y -----------
// grid(64 l-tiles, 4 co-tiles, 8)
__global__ void k_final(const __bf16* __restrict__ yb, const __bf16* __restrict__ xb,
                        const __bf16* __restrict__ wgb,  // [256][512]
                        const __bf16* __restrict__ wrb,  // [256][256]
                        const float* __restrict__ y, float* __restrict__ out) {
  __shared__ __align__(16) __bf16 Asg[64 * 32];
  __shared__ __align__(16) __bf16 Asr[64 * 32];
  __shared__ __align__(16) __bf16 BsRaw[66 * BSTRIDE];
  __bf16* Bs = BsRaw + BSTRIDE;
  const int t = threadIdx.x, lane = t & 31, wav = t >> 5;
  const int wm = wav >> 1, wn = wav & 1;
  const int l0 = blockIdx.x * 64, c0 = blockIdx.y * 64, b = blockIdx.z;
  const __bf16* py = yb + (size_t)b * 256 * LL;
  const __bf16* px = xb + (size_t)b * 256 * LL;
  const int mA = t >> 2, kA = (t & 3) * 8;
  const int kkB = t >> 3, nbB = (t & 7) * 8;
  v8f ag0 = {}, ag1 = {}, ar0 = {}, ar1 = {};

  // phase 1: gate += Wg[:,0:256] * y
  v8bf bReg = *(const v8bf*)(py + ((size_t)kkB << 12) + l0 + nbB);
  for (int k0 = 0; k0 < 256; k0 += 32) {
    __syncthreads();
    async_copy16(wgb + (size_t)(c0 + mA) * 512 + k0 + kA, Asg + mA * 32 + kA);
#pragma unroll
    for (int i = 0; i < 8; ++i) Bs[(nbB + i) * BSTRIDE + kkB] = bReg[i];
    WAIT_ASYNC();
    __syncthreads();
    const int kn = k0 + 32;
    if (kn < 256)
      bReg = *(const v8bf*)(py + ((size_t)(kn + kkB) << 12) + l0 + nbB);
    v16bf af = load_afrag(Asg + wm * 16 * 32, lane);
    ag0 = wmma_bf16(af, load_bfrag(Bs + (wn * 32) * BSTRIDE, lane), ag0);
    ag1 = wmma_bf16(af, load_bfrag(Bs + (wn * 32 + 16) * BSTRIDE, lane), ag1);
  }
  // phase 2: gate += Wg[:,256:512] * x ; res = Wr * x (shared B tile)
  bReg = *(const v8bf*)(px + ((size_t)kkB << 12) + l0 + nbB);
  for (int k0 = 0; k0 < 256; k0 += 32) {
    __syncthreads();
    async_copy16(wgb + (size_t)(c0 + mA) * 512 + 256 + k0 + kA, Asg + mA * 32 + kA);
    async_copy16(wrb + (size_t)(c0 + mA) * 256 + k0 + kA, Asr + mA * 32 + kA);
#pragma unroll
    for (int i = 0; i < 8; ++i) Bs[(nbB + i) * BSTRIDE + kkB] = bReg[i];
    WAIT_ASYNC();
    __syncthreads();
    const int kn = k0 + 32;
    if (kn < 256)
      bReg = *(const v8bf*)(px + ((size_t)(kn + kkB) << 12) + l0 + nbB);
    v16bf bf0 = load_bfrag(Bs + (wn * 32) * BSTRIDE, lane);
    v16bf bf1 = load_bfrag(Bs + (wn * 32 + 16) * BSTRIDE, lane);
    v16bf afg = load_afrag(Asg + wm * 16 * 32, lane);
    v16bf afr = load_afrag(Asr + wm * 16 * 32, lane);
    ag0 = wmma_bf16(afg, bf0, ag0);
    ag1 = wmma_bf16(afg, bf1, ag1);
    ar0 = wmma_bf16(afr, bf0, ar0);
    ar1 = wmma_bf16(afr, bf1, ar1);
  }
  const int h = lane >> 4, nI = lane & 15;
#pragma unroll
  for (int j = 0; j < 2; ++j) {
    v8f ag = j ? ag1 : ag0;
    v8f ar = j ? ar1 : ar0;
    const int l = l0 + wn * 32 + j * 16 + nI;
#pragma unroll
    for (int r = 0; r < 8; ++r) {
      int co = c0 + wm * 16 + h * 8 + r;
      size_t off = ((size_t)b * 256 + co) * LL + l;
      float g = 1.0f / (1.0f + __expf(-ag[r]));
      out[off] = g * ar[r] + (1.0f - g) * y[off];
    }
  }
}

// ------------------------------- launcher -----------------------------------

extern "C" void kernel_launch(void* const* d_in, const int* in_sizes, int n_in,
                              void* d_out, int out_size, void* d_ws, size_t ws_size,
                              hipStream_t stream) {
  const float* x      = (const float*)d_in[0];
  const float* w_qkv  = (const float*)d_in[1];
  const float* w_mod  = (const float*)d_in[2];
  const float* w_res  = (const float*)d_in[3];
  const float* w_gate = (const float*)d_in[4];
  float* out = (float*)d_out;

  constexpr size_t NE = (size_t)BQ * 256 * LL;
  char* ws = (char*)d_ws;
  size_t off = 0;
  auto alloc = [&](size_t bytes) { char* p = ws + off; off += (bytes + 255) & ~(size_t)255; return p; };

  __bf16* xb    = (__bf16*)alloc(NE * 2);
  __bf16* wqkvb = (__bf16*)alloc((size_t)9 * 768 * 256 * 2);
  __bf16* wmodb = (__bf16*)alloc((size_t)3 * 512 * 256 * 2);
  __bf16* wgb   = (__bf16*)alloc((size_t)256 * 512 * 2);
  __bf16* wrb   = (__bf16*)alloc((size_t)256 * 256 * 2);
  float*  q     = (float*)alloc(NE * 4);
  float*  kk    = (float*)alloc(NE * 4);
  __bf16* vb    = (__bf16*)alloc(NE * 2);
  float*  qm    = (float*)alloc(NE * 4);
  float*  km    = (float*)alloc(NE * 4);
  __bf16* mqb   = (__bf16*)alloc(NE * 2);
  __bf16* mkb   = (__bf16*)alloc(NE * 2);
  float*  S     = (float*)alloc((size_t)BQ * 256 * 256 * 4);
  __bf16* mab   = (__bf16*)alloc((size_t)BQ * 256 * 256 * 2);
  float*  y     = (float*)alloc(NE * 4);
  __bf16* yb    = (__bf16*)alloc(NE * 2);
  if (off > ws_size) return;

  k_cast_bf16<<<dim3((NE + 255) / 256), 256, 0, stream>>>(x, xb, (int)NE);
  k_cast_wqkv<<<dim3((768 * 256 * 9 + 255) / 256), 256, 0, stream>>>(w_qkv, wqkvb);
  k_cast_wmod<<<dim3((512 * 256 * 3 + 255) / 256), 256, 0, stream>>>(w_mod, wmodb);
  k_cast_bf16<<<dim3((256 * 512 + 255) / 256), 256, 0, stream>>>(w_gate, wgb, 256 * 512);
  k_cast_bf16<<<dim3((256 * 256 + 255) / 256), 256, 0, stream>>>(w_res, wrb, 256 * 256);

  k_conv_qkv<<<dim3(64, 12, BQ), 256, 0, stream>>>(xb, wqkvb, q, kk, vb);
  k_conv_mod<<<dim3(64, 8, BQ), 256, 0, stream>>>(vb, wmodb, qm, km);
  k_mod_softmax<<<dim3(BQ * 256), 256, 0, stream>>>(q, qm, mqb);
  k_mod_softmax<<<dim3(BQ * 256), 256, 0, stream>>>(kk, km, mkb);
  k_attn_scores<<<dim3(4, 4, BQ), 256, 0, stream>>>(mqb, mkb, S);
  k_attn_softmax<<<dim3(BQ * 256), 256, 0, stream>>>(S, mab);
  k_attn_apply<<<dim3(64, 4, BQ), 256, 0, stream>>>(mab, vb, y, yb);
  k_final<<<dim3(64, 4, BQ), 256, 0, stream>>>(yb, xb, wgb, wrb, y, out);
}
